// PDFormer_73770358276503
// MI455X (gfx1250) — matrix operand order repaired and checked by reading, not compiled, not run on hardware
//
#include <hip/hip_runtime.h>
#include <hip/hip_bf16.h>

typedef __attribute__((ext_vector_type(16))) __bf16 v16bf;
typedef __attribute__((ext_vector_type(8)))  float  v8f;
typedef unsigned int v4u __attribute__((ext_vector_type(4)));
typedef int          v8i __attribute__((ext_vector_type(8)));
typedef int          v4i __attribute__((ext_vector_type(4)));

#define B_ 4
#define T_ 12
#define N_ 640
#define D_ 128
#define NTOK (B_ * T_ * N_)   // 30720
#define PC 448                // concatenated projection columns
#define SS 644                // LDS score stride (bank-conflict-free)

#if __has_builtin(__builtin_amdgcn_tensor_load_to_lds) && \
    __has_builtin(__builtin_amdgcn_s_wait_tensorcnt)
#define USE_TDM 1
#endif

// ---- WMMA lane-layout helpers (CDNA5 ISA 7.12.2, wave32) ----
// A matrix 16x32 bf16: lane = (halfsel<<4)|row ; element e -> K index:
__device__ __forceinline__ int a_kidx(int e, int hs) {
    int p = e >> 1, w = e & 1;
    int base = (p < 4) ? (p * 2) : (16 + (p - 4) * 2);
    return base + hs * 8 + w;
}

__device__ __forceinline__ v8f wmma_bf(v16bf a, v16bf b, v8f c) {
    return __builtin_amdgcn_wmma_f32_16x16x32_bf16(false, a, false, b,
                                                   (short)0, c, false, false);
}

// ============================================================
// Kernel 0: weight prep (Wcat bf16 [448x128], Wproj bf16, pk/pv f32 [16x64])
// Wcat rows: gq(0) gk(64) gv(128) sq(192) sk(224) sv(256) tq(288) tk(320) tv(352) pq(384)
// ============================================================
__global__ void k_prep(const float* __restrict__ pattern_keys,
                       const float* __restrict__ t_q_w, const float* __restrict__ t_k_w,
                       const float* __restrict__ t_v_w,
                       const float* __restrict__ geo_q_w, const float* __restrict__ geo_k_w,
                       const float* __restrict__ geo_v_w,
                       const float* __restrict__ sem_q_w, const float* __restrict__ sem_k_w,
                       const float* __restrict__ sem_v_w,
                       const float* __restrict__ pat_q_w,
                       const float* __restrict__ pat_k_w, const float* __restrict__ pat_k_b,
                       const float* __restrict__ pat_v_w, const float* __restrict__ pat_v_b,
                       const float* __restrict__ proj_w,
                       float* __restrict__ pk, float* __restrict__ pv,
                       __bf16* __restrict__ Wcat, __bf16* __restrict__ Wproj) {
    int idx = blockIdx.x * blockDim.x + threadIdx.x;
    if (idx < 448 * 128) {
        int r = idx >> 7, d = idx & 127;
        float v;
        if      (r < 64)  v = geo_q_w[r * 128 + d];
        else if (r < 128) v = geo_k_w[(r - 64) * 128 + d];
        else if (r < 192) v = geo_v_w[(r - 128) * 128 + d];
        else if (r < 224) v = sem_q_w[(r - 192) * 128 + d];
        else if (r < 256) v = sem_k_w[(r - 224) * 128 + d];
        else if (r < 288) v = sem_v_w[(r - 256) * 128 + d];
        else if (r < 320) v = t_q_w[(r - 288) * 128 + d];
        else if (r < 352) v = t_k_w[(r - 320) * 128 + d];
        else if (r < 384) v = t_v_w[(r - 352) * 128 + d];
        else              v = pat_q_w[(r - 384) * 128 + d];
        Wcat[idx] = (__bf16)v;
        return;
    }
    int i2 = idx - 448 * 128;
    if (i2 < 128 * 128) { Wproj[i2] = (__bf16)proj_w[i2]; return; }
    int i3 = i2 - 128 * 128;
    if (i3 < 1024) {
        int k = i3 >> 6, e = i3 & 63;
        float s = 0.f;
        for (int d = 0; d < 128; ++d) s += pattern_keys[k * 128 + d] * pat_k_w[e * 128 + d];
        pk[i3] = s + pat_k_b[e];
        return;
    }
    int i4 = i3 - 1024;
    if (i4 < 1024) {
        int k = i4 >> 6, e = i4 & 63;
        float s = 0.f;
        for (int d = 0; d < 128; ++d) s += pattern_keys[k * 128 + d] * pat_v_w[e * 128 + d];
        pv[i4] = s + pat_v_b[e];
    }
}

// ============================================================
// Kernel 1: fused projection GEMM  P[30720 x 448] (bf16)
//   cols 0..383 <- x @ Wcat^T ; cols 384..447 <- x_patterns @ pat_q_w^T
// 256 threads (8 waves), 128 tokens per block.
// x / x_patterns tiles staged into LDS (padded stride 132 floats) — via the
// Tensor Data Mover (HW-applied row padding) when available.
// ============================================================
__global__ void k_proj(const float* __restrict__ x, const float* __restrict__ xp,
                       const __bf16* __restrict__ Wcat, __bf16* __restrict__ P) {
    extern __shared__ float lds1[];
    float* Xs  = lds1;                 // 128 x 132 (padded)
    float* XPs = lds1 + 128 * 132;
    const int tid = threadIdx.x;
    const int token0 = blockIdx.x * 128;

    if (tid < 112) __builtin_prefetch(Wcat + tid * 512, 0, 1);  // global_prefetch

#ifdef USE_TDM
    if (tid < 32) {
        // D# group1: data_size=4B, pad_enable, pad_interval=128 dwords (code 6),
        // pad_amount=4 dwords (code 3) -> LDS row stride 132 floats.
        // tensor: dim0=128, dim0_stride=128, dim1=NTOK; tile 128x128.
        const v8i g1 = { (int)((2u << 16) | (1u << 20) | (6u << 22) | (3u << 25)),
                         (int)(128u << 16),          // tensor_dim0 = 128 (bits 79:48 lo)
                         (int)(30720u << 16),        // tensor_dim1 = 30720 (bits 111:80 lo)
                         (int)(128u << 16),          // tile_dim0 = 128 (bits 127:112)
                         128,                        // tile_dim1 = 128 (bits 143:128)
                         128,                        // tensor_dim0_stride = 128 (bits 191:160)
                         0, 0 };
        const v4i z4 = { 0, 0, 0, 0 };
        const v8i z8 = { 0, 0, 0, 0, 0, 0, 0, 0 };
        unsigned long long gx = (unsigned long long)(size_t)(const void*)(x  + (size_t)token0 * 128);
        unsigned long long gp = (unsigned long long)(size_t)(const void*)(xp + (size_t)token0 * 128);
        v4u g0x = { 1u, 0u,
                    (unsigned)(gx & 0xFFFFFFFFu),
                    (unsigned)((gx >> 32) & 0x1FFFFFFu) | (2u << 30) };
        v4u g0p = { 1u, 67584u,   // lds_addr of XPs (bytes)
                    (unsigned)(gp & 0xFFFFFFFFu),
                    (unsigned)((gp >> 32) & 0x1FFFFFFu) | (2u << 30) };
        __builtin_amdgcn_tensor_load_to_lds(g0x, g1, z4, z4, z8, 0);
        __builtin_amdgcn_tensor_load_to_lds(g0p, g1, z4, z4, z8, 0);
        __builtin_amdgcn_s_wait_tensorcnt(0);
    }
#else
    for (int i = tid; i < 128 * 32; i += 256) {
        int row = i >> 5, c4 = (i & 31) * 4;
        float4 vx = *(const float4*)(x  + (size_t)(token0 + row) * 128 + c4);
        float4 vp = *(const float4*)(xp + (size_t)(token0 + row) * 128 + c4);
        *(float4*)&Xs [row * 132 + c4] = vx;
        *(float4*)&XPs[row * 132 + c4] = vp;
    }
#endif
    __syncthreads();

    const int lane = tid & 31, wave = tid >> 5;
    const int arow = lane & 15, hs = lane >> 4;
    const int bcol = lane & 15, kb = hs * 16;
    int kidx[16];
#pragma unroll
    for (int e = 0; e < 16; ++e) kidx[e] = a_kidx(e, hs);

    // hoist A operands: this wave owns row-tile == wave
    v16bf ax[4], apt[4];
#pragma unroll
    for (int kk = 0; kk < 4; ++kk) {
#pragma unroll
        for (int e = 0; e < 16; ++e) {
            int off = (wave * 16 + arow) * 132 + kk * 32 + kidx[e];
            ax[kk][e]  = (__bf16)Xs[off];
            apt[kk][e] = (__bf16)XPs[off];
        }
    }

    // ct 0..23: x-projection columns (A = ax)
    for (int ct = 0; ct < 24; ++ct) {
        v16bf b[4];
#pragma unroll
        for (int kk = 0; kk < 4; ++kk)
#pragma unroll
            for (int e = 0; e < 16; ++e)
                b[kk][e] = Wcat[(ct * 16 + bcol) * 128 + kk * 32 + kidx[e]];
        v8f c;
#pragma unroll
        for (int r = 0; r < 8; ++r) c[r] = 0.f;
#pragma unroll
        for (int kk = 0; kk < 4; ++kk) c = wmma_bf(ax[kk], b[kk], c);
#pragma unroll
        for (int r = 0; r < 8; ++r)
            P[(size_t)(token0 + wave * 16 + hs * 8 + r) * PC + ct * 16 + bcol] = (__bf16)c[r];
    }
    // ct 24..27: pattern-query columns (A = apt)
    for (int ct = 24; ct < 28; ++ct) {
        v16bf b[4];
#pragma unroll
        for (int kk = 0; kk < 4; ++kk)
#pragma unroll
            for (int e = 0; e < 16; ++e)
                b[kk][e] = Wcat[(ct * 16 + bcol) * 128 + kk * 32 + kidx[e]];
        v8f c;
#pragma unroll
        for (int r = 0; r < 8; ++r) c[r] = 0.f;
#pragma unroll
        for (int kk = 0; kk < 4; ++kk) c = wmma_bf(apt[kk], b[kk], c);
#pragma unroll
        for (int r = 0; r < 8; ++r)
            P[(size_t)(token0 + wave * 16 + hs * 8 + r) * PC + ct * 16 + bcol] = (__bf16)c[r];
    }
}

// ============================================================
// Kernel 2: pattern attention, gk += softmax(pq pk^T * s) pv   (in-place on P)
// ============================================================
__global__ void k_pattern(__bf16* __restrict__ P, const float* __restrict__ pk,
                          const float* __restrict__ pv, const float* __restrict__ pat_q_b) {
    int t = blockIdx.x * blockDim.x + threadIdx.x;
    if (t >= NTOK) return;
    __bf16* row = P + (size_t)t * PC;
    float pq[64];
#pragma unroll
    for (int e = 0; e < 64; ++e) pq[e] = (float)row[384 + e] + pat_q_b[e];
    float lg[16], mx = -3.4e38f;
    for (int k = 0; k < 16; ++k) {
        float s = 0.f;
        for (int e = 0; e < 64; ++e) s += pq[e] * pk[k * 64 + e];
        s *= 0.25f;
        lg[k] = s;
        mx = fmaxf(mx, s);
    }
    float sum = 0.f;
    for (int k = 0; k < 16; ++k) { lg[k] = __expf(lg[k] - mx); sum += lg[k]; }
    float inv = 1.f / sum;
    for (int e = 0; e < 64; ++e) {
        float acc = (float)row[64 + e];
        for (int k = 0; k < 16; ++k) acc += lg[k] * inv * pv[k * 64 + e];
        row[64 + e] = (__bf16)acc;
    }
}

// ============================================================
// Kernel 3: temporal attention over T=12 per (b,n,head); writes Y cols 0..31
// ============================================================
__global__ void k_temporal(const __bf16* __restrict__ P, __bf16* __restrict__ Y) {
    int idx = blockIdx.x * blockDim.x + threadIdx.x;
    if (idx >= B_ * N_ * 2 * T_) return;
    int i = idx % T_;  int r = idx / T_;
    int h = r & 1;     r >>= 1;
    int n = r % N_;    int b = r / N_;

    float q[16];
    size_t bi = ((size_t)(b * T_ + i) * N_ + n) * PC;
#pragma unroll
    for (int e = 0; e < 16; ++e) q[e] = (float)P[bi + 288 + h * 16 + e];

    float lg[T_], mx = -3.4e38f;
    for (int j = 0; j < T_; ++j) {
        size_t bj = ((size_t)(b * T_ + j) * N_ + n) * PC;
        float s = 0.f;
#pragma unroll
        for (int e = 0; e < 16; ++e) s += q[e] * (float)P[bj + 320 + h * 16 + e];
        lg[j] = s * 0.25f;
        mx = fmaxf(mx, lg[j]);
    }
    float sum = 0.f;
    for (int j = 0; j < T_; ++j) { lg[j] = __expf(lg[j] - mx); sum += lg[j]; }
    float inv = 1.f / sum;

    float o[16];
#pragma unroll
    for (int e = 0; e < 16; ++e) o[e] = 0.f;
    for (int j = 0; j < T_; ++j) {
        size_t bj = ((size_t)(b * T_ + j) * N_ + n) * PC;
        float p = lg[j] * inv;
#pragma unroll
        for (int e = 0; e < 16; ++e) o[e] += p * (float)P[bj + 352 + h * 16 + e];
    }
    size_t yo = ((size_t)(b * T_ + i) * N_ + n) * (size_t)D_ + h * 16;
#pragma unroll
    for (int e = 0; e < 16; ++e) Y[yo + e] = (__bf16)o[e];
}

// ============================================================
// Kernel 4: spatial attention (geo 4 heads + sem 2 heads) over N=640
// grid (48 bt, 10 qtiles of 64), 128 threads (4 waves). Each wave handles
// (head, 16-query) tasks; per-wave 16x644 f32 score strip in LDS.
// ============================================================
__global__ void k_spatial(const __bf16* __restrict__ P, __bf16* __restrict__ Y,
                          const unsigned char* __restrict__ geo_mask,
                          const unsigned char* __restrict__ sem_mask) {
    extern __shared__ float lds4[];
    const int bt = blockIdx.x;     // b*T + t
    const int qt = blockIdx.y;     // 64-query tile
    const int tid = threadIdx.x, lane = tid & 31, wave = tid >> 5;
    float* sc = lds4 + wave * 16 * SS;

    const int arow = lane & 15, hs = lane >> 4, mb = hs * 8;
    const int bcol = lane & 15, kb = hs * 16;
    int kidx[16];
#pragma unroll
    for (int e = 0; e < 16; ++e) kidx[e] = a_kidx(e, hs);
    const size_t tokbase = (size_t)bt * N_;

    for (int task = wave; task < 24; task += 4) {
        const int hid = task >> 2, sub = task & 3;
        int qcol, kcol, vcol, ocol;
        const unsigned char* mask;
        if (hid < 4) {
            qcol = hid * 16; kcol = 64 + hid * 16; vcol = 128 + hid * 16;
            ocol = 32 + hid * 16; mask = geo_mask;
        } else {
            int s = hid - 4;
            qcol = 192 + s * 16; kcol = 224 + s * 16; vcol = 256 + s * 16;
            ocol = 96 + s * 16; mask = sem_mask;
        }
        const int q0 = qt * 64 + sub * 16;

        // A = Q tile, K-dim 16 zero-padded to 32
        v16bf aq;
#pragma unroll
        for (int e = 0; e < 16; ++e) {
            int k = kidx[e];
            aq[e] = (k < 16) ? P[(tokbase + q0 + arow) * PC + qcol + k] : (__bf16)0.f;
        }

        // Phase 1: S = Q K^T * scale, masked, into LDS
        for (int kc = 0; kc < N_; kc += 16) {
            v16bf bk;
#pragma unroll
            for (int e = 0; e < 16; ++e) {
                int k = kidx[e];
                bk[e] = (k < 16) ? P[(tokbase + kc + bcol) * PC + kcol + k] : (__bf16)0.f;
            }
            v8f c;
#pragma unroll
            for (int r = 0; r < 8; ++r) c[r] = 0.f;
            c = wmma_bf(aq, bk, c);
#pragma unroll
            for (int r = 0; r < 8; ++r) {
                int m = mb + r, key = kc + bcol;
                float v = c[r] * 0.25f;
                if (mask[(size_t)(q0 + m) * N_ + key]) v = -3.0e38f;
                sc[m * SS + key] = v;
            }
        }

        // Softmax: lanes 0..15 each own one row (same-wave LDS ops are in-order)
        if (lane < 16) {
            float* row = sc + lane * SS;
            float mxv = -3.4e38f;
            for (int j = 0; j < N_; ++j) mxv = fmaxf(mxv, row[j]);
            float s = 0.f;
            for (int j = 0; j < N_; ++j) { float e = __expf(row[j] - mxv); row[j] = e; s += e; }
            float inv = 1.f / s;
            for (int j = 0; j < N_; ++j) row[j] *= inv;
        }

        // Phase 2: O = P V
        v8f co;
#pragma unroll
        for (int r = 0; r < 8; ++r) co[r] = 0.f;
        for (int kc = 0; kc < N_; kc += 32) {
            v16bf ap, bv;
#pragma unroll
            for (int e = 0; e < 16; ++e)
                ap[e] = (__bf16)sc[arow * SS + kc + kidx[e]];
#pragma unroll
            for (int e = 0; e < 16; ++e)
                bv[e] = P[(tokbase + kc + kidx[e]) * PC + vcol + bcol];
            co = wmma_bf(ap, bv, co);
        }
#pragma unroll
        for (int r = 0; r < 8; ++r)
            Y[(tokbase + q0 + mb + r) * (size_t)D_ + ocol + bcol] = (__bf16)co[r];
    }
}

// ============================================================
// Kernel 5: output projection  out = Y @ proj_w^T + proj_b  (f32 out)
// ============================================================
__global__ void k_out(const __bf16* __restrict__ Y, const __bf16* __restrict__ Wp,
                      const float* __restrict__ proj_b, float* __restrict__ out) {
    const int tid = threadIdx.x;
    const int token0 = blockIdx.x * 128;
    const int lane = tid & 31, wave = tid >> 5;
    const int arow = lane & 15, hs = lane >> 4;
    const int bcol = lane & 15, kb = hs * 16;
    int kidx[16];
#pragma unroll
    for (int e = 0; e < 16; ++e) kidx[e] = a_kidx(e, hs);

    v16bf a[4];
#pragma unroll
    for (int kk = 0; kk < 4; ++kk) {
#pragma unroll
        for (int e = 0; e < 16; ++e)
            a[kk][e] = Y[(size_t)(token0 + wave * 16 + arow) * D_ + kk * 32 + kidx[e]];
    }
    for (int ct = 0; ct < 8; ++ct) {
        v16bf b[4];
#pragma unroll
        for (int kk = 0; kk < 4; ++kk)
#pragma unroll
            for (int e = 0; e < 16; ++e)
                b[kk][e] = Wp[(ct * 16 + bcol) * 128 + kk * 32 + kidx[e]];
        v8f c;
#pragma unroll
        for (int r = 0; r < 8; ++r) c[r] = 0.f;
#pragma unroll
        for (int kk = 0; kk < 4; ++kk) c = wmma_bf(a[kk], b[kk], c);
        float bias = proj_b[ct * 16 + bcol];
#pragma unroll
        for (int r = 0; r < 8; ++r)
            out[(size_t)(token0 + wave * 16 + hs * 8 + r) * D_ + ct * 16 + bcol] = c[r] + bias;
    }
}

// ============================================================
extern "C" void kernel_launch(void* const* d_in, const int* in_sizes, int n_in,
                              void* d_out, int out_size, void* d_ws, size_t ws_size,
                              hipStream_t stream) {
    (void)in_sizes; (void)n_in; (void)out_size; (void)ws_size;
    const float* x            = (const float*)d_in[0];
    const float* xp           = (const float*)d_in[1];
    const float* pattern_keys = (const float*)d_in[2];
    const unsigned char* geo_mask = (const unsigned char*)d_in[3];
    const unsigned char* sem_mask = (const unsigned char*)d_in[4];
    const float* t_q_w   = (const float*)d_in[5];
    const float* t_k_w   = (const float*)d_in[6];
    const float* t_v_w   = (const float*)d_in[7];
    const float* geo_q_w = (const float*)d_in[8];
    const float* geo_k_w = (const float*)d_in[9];
    const float* geo_v_w = (const float*)d_in[10];
    const float* sem_q_w = (const float*)d_in[11];
    const float* sem_k_w = (const float*)d_in[12];
    const float* sem_v_w = (const float*)d_in[13];
    const float* pat_q_w = (const float*)d_in[14];
    const float* pat_q_b = (const float*)d_in[15];
    const float* pat_k_w = (const float*)d_in[16];
    const float* pat_k_b = (const float*)d_in[17];
    const float* pat_v_w = (const float*)d_in[18];
    const float* pat_v_b = (const float*)d_in[19];
    const float* proj_w  = (const float*)d_in[20];
    const float* proj_b  = (const float*)d_in[21];
    float* out = (float*)d_out;

    // workspace carve-up (~34 MB)
    char* w = (char*)d_ws;
    float*  pk    = (float*)w;  w += 16 * 64 * sizeof(float);
    float*  pv    = (float*)w;  w += 16 * 64 * sizeof(float);
    __bf16* Wcat  = (__bf16*)w; w += 448 * 128 * sizeof(__bf16);
    __bf16* Wproj = (__bf16*)w; w += 128 * 128 * sizeof(__bf16);
    __bf16* Pbuf  = (__bf16*)w; w += (size_t)NTOK * PC * sizeof(__bf16);
    __bf16* Ybuf  = (__bf16*)w; w += (size_t)NTOK * D_ * sizeof(__bf16);

    k_prep<<<(448 * 128 + 128 * 128 + 2048 + 255) / 256, 256, 0, stream>>>(
        pattern_keys, t_q_w, t_k_w, t_v_w, geo_q_w, geo_k_w, geo_v_w,
        sem_q_w, sem_k_w, sem_v_w, pat_q_w, pat_k_w, pat_k_b, pat_v_w, pat_v_b,
        proj_w, pk, pv, Wcat, Wproj);

    size_t lds1 = 2 * 128 * 132 * sizeof(float);   // 135168 B
    k_proj<<<NTOK / 128, 256, lds1, stream>>>(x, xp, Wcat, Pbuf);

    k_pattern<<<NTOK / 256, 256, 0, stream>>>(Pbuf, pk, pv, pat_q_b);

    k_temporal<<<(B_ * N_ * 2 * T_) / 256, 256, 0, stream>>>(Pbuf, Ybuf);

    size_t lds4 = 4 * 16 * SS * sizeof(float);     // 164864 B
    dim3 g4(B_ * T_, N_ / 64);
    k_spatial<<<g4, 128, lds4, stream>>>(Pbuf, Ybuf, geo_mask, sem_mask);

    k_out<<<NTOK / 128, 256, 0, stream>>>(Ybuf, Wproj, proj_b, out);
}
